// RelPositionMultiHeadAttentionLongformer_21663815041771
// MI455X (gfx1250) — matrix operand order repaired and checked by reading, not compiled, not run on hardware
//
#include <hip/hip_runtime.h>
#include <hip/hip_bf16.h>
#include <stdint.h>

// ---------------------------------------------------------------------------
// Longformer rel-pos attention for MI455X (gfx1250), wave32, WMMA bf16 + TDM.
// Pipeline:
//   gemm<1>: qu/qv = (x@Wq^T + bq) + pos_bias_{u,v}   -> bf16 (B,H,T,DK)
//   gemm<0>: k, v projections                          -> bf16 (B,H,T,DK)
//   pos_proj: p = pos_emb@Wp^T                         -> bf16 (H,272,DK)
//   attn_band_wmma: banded scores + softmax + AV       -> fp32 ctx (B,T,NF)
//                   (contiguous LDS staging via tensor_load_to_lds / TENSORcnt)
//   gemm<2>: out = ctx@Wo^T + bo                       -> fp32 d_out
// ---------------------------------------------------------------------------

#define H_     8
#define DK_    64
#define NF_    512
#define T_     4096
#define B_     4
#define W_     128
#define BAND_  257
#define BANDP_ 272          // band padded to 17 tiles of 16
#define WIN3_  384          // 3*W window
#define NEG_   (-10000.0f)

typedef __attribute__((ext_vector_type(16))) __bf16 v16bf;
typedef __attribute__((ext_vector_type(8)))  __bf16 v8bf;
typedef __attribute__((ext_vector_type(8)))  float  v8f;
typedef __attribute__((ext_vector_type(4)))  unsigned int u32x4;
typedef __attribute__((ext_vector_type(8)))  int i32x8;
typedef __attribute__((ext_vector_type(4)))  int i32x4;

#if defined(__HIP_DEVICE_COMPILE__) && \
    __has_builtin(__builtin_amdgcn_tensor_load_to_lds) && \
    __has_builtin(__builtin_amdgcn_s_wait_tensorcnt)
#define USE_TDM 1
#else
#define USE_TDM 0
#endif

__device__ __forceinline__ v8f v8f_zero() {
  v8f z;
#pragma unroll
  for (int i = 0; i < 8; ++i) z[i] = 0.0f;
  return z;
}

#if USE_TDM
// Issue one TDM 1D contiguous copy (global -> LDS), data_size = 8B units.
// D# packing per CDNA5 ISA 8.3/8.4: group0 = {count=1, lds_addr, global_addr,
// type=2}; group1 = {data_size=3, tensor_dim0 = stride = tile_dim0 = units}.
// Must be executed by exactly one wave (EXEC is ignored; one instr = one DMA).
__device__ __forceinline__ void tdm_load_1d(unsigned lds_off,
                                            const void* gptr,
                                            unsigned bytes) {
  const unsigned units = bytes >> 3;                 // 8-byte elements
  const unsigned long long ga = (unsigned long long)(uintptr_t)gptr;
  u32x4 g0;
  g0[0] = 1u;                                        // count=1, user mode
  g0[1] = lds_off;                                   // LDS byte address
  g0[2] = (unsigned)(ga & 0xFFFFFFFFu);              // global_addr[31:0]
  g0[3] = (unsigned)((ga >> 32) & 0x1FFFFFFu)        // global_addr[56:32]
        | 0x80000000u;                               // type=2 ("image")
  i32x8 g1;
  g1[0] = (int)(3u << 16);                           // data_size=3 (8B)
  g1[1] = (int)((units & 0xFFFFu) << 16);            // tensor_dim0[15:0]
  g1[2] = (int)(((units >> 16) & 0xFFFFu)            // tensor_dim0[31:16]
        | (1u << 16));                               // tensor_dim1 = 1
  g1[3] = (int)((units & 0xFFFFu) << 16);            // tile_dim0
  g1[4] = 0;                                         // tile_dim1/2 unused
  g1[5] = (int)units;                                // tensor_dim0_stride lo
  g1[6] = 0;
  g1[7] = 0;
  i32x4 gz;
  gz[0] = 0; gz[1] = 0; gz[2] = 0; gz[3] = 0;        // groups 2/3 unused
#if __clang_major__ >= 23
  i32x8 g4;
#pragma unroll
  for (int i = 0; i < 8; ++i) g4[i] = 0;
  __builtin_amdgcn_tensor_load_to_lds(g0, g1, gz, gz, g4, 0);
#else
  __builtin_amdgcn_tensor_load_to_lds(g0, g1, gz, gz, 0);
#endif
}
#endif

// A fragment (16x32 bf16, contiguous K to match frag_b):
//  lanes 0-15 : row=lane,   v0..7 = K0..15
//  lanes 16-31: row=lane-16,v0..7 = K16..31
// LDS source is plain row-major (row, 32 contiguous K), `stride` in elements.
__device__ __forceinline__ v16bf frag_a(const __bf16* base, int stride) {
  const int lane = threadIdx.x & 31;
  const int r = lane & 15, half = lane >> 4;
  const __bf16* p = base + r * stride + half * 16;
  v8bf lo = *(const v8bf*)(p);
  v8bf hi = *(const v8bf*)(p + 8);
  v16bf out;
#pragma unroll
  for (int i = 0; i < 8; ++i) { out[i] = lo[i]; out[i + 8] = hi[i]; }
  return out;
}

// B fragment (32x16 bf16): lane n=lane&15 holds column n; lanes 0-15 K0..15,
// lanes 16-31 K16..31 (contiguous K, same order as frag_a).
// LDS source stored as (N rows, K contiguous) -> one 32B contiguous load.
__device__ __forceinline__ v16bf frag_b(const __bf16* base, int stride) {
  const int lane = threadIdx.x & 31;
  const int r = lane & 15, half = lane >> 4;
  return *(const v16bf*)(base + r * stride + half * 16);
}

__device__ __forceinline__ v8f wmma32(v16bf a, v16bf b, v8f c) {
  return __builtin_amdgcn_wmma_f32_16x16x32_bf16(false, a, false, b,
                                                 (short)0, c, false, false);
}

// ---------------------------------------------------------------------------
// Tiled GEMM: C[M=16384, N=512] = A[M,512] @ Wm[N,512]^T (+ bias epilogues)
// MODE 0: out0[b,h,t,d] = bf16(c + bias[n])                       (k or v)
// MODE 1: out0 = bf16(c + bias + pbu[n]); out1 = bf16(c+bias+pbv) (q_u,q_v)
// MODE 2: out2[m,n] = c + bias[n]                                 (final fp32)
// ---------------------------------------------------------------------------
template <int MODE>
__global__ __launch_bounds__(256)
void gemm512_wmma(const float* __restrict__ A, const float* __restrict__ Wm,
                  const float* __restrict__ bias,
                  __bf16* __restrict__ out0, __bf16* __restrict__ out1,
                  float* __restrict__ out2,
                  const float* __restrict__ pbu, const float* __restrict__ pbv) {
  __shared__ __bf16 sA[128 * 32];
  __shared__ __bf16 sB[128 * 32];
  const int tid = threadIdx.x;
  const int wv = tid >> 5, lane = tid & 31;
  const int m0 = blockIdx.y * 128, n0 = blockIdx.x * 128;

  v8f acc[8];
#pragma unroll
  for (int i = 0; i < 8; ++i) acc[i] = v8f_zero();

  for (int k0 = 0; k0 < 512; k0 += 32) {
#pragma unroll
    for (int q = 0; q < 4; ++q) {
      int u = tid + 256 * q;                 // 0..1023 float4 slots
      int row = u >> 3, c4 = (u & 7) * 4;
      float4 fa = *(const float4*)(A + (size_t)(m0 + row) * 512 + k0 + c4);
      __bf16* da = sA + row * 32 + c4;
      da[0] = (__bf16)fa.x; da[1] = (__bf16)fa.y;
      da[2] = (__bf16)fa.z; da[3] = (__bf16)fa.w;
      float4 fb = *(const float4*)(Wm + (size_t)(n0 + row) * 512 + k0 + c4);
      __bf16* db = sB + row * 32 + c4;
      db[0] = (__bf16)fb.x; db[1] = (__bf16)fb.y;
      db[2] = (__bf16)fb.z; db[3] = (__bf16)fb.w;
    }
    __syncthreads();
    v16bf af = frag_a(sA + wv * 16 * 32, 32);
#pragma unroll
    for (int nt = 0; nt < 8; ++nt) {
      v16bf bf = frag_b(sB + nt * 16 * 32, 32);
      acc[nt] = wmma32(af, bf, acc[nt]);
    }
    __syncthreads();
  }

  // epilogue: element (m,n): m = m0+16w+8*(lane>>4)+v, n = n0+16*nt+(lane&15)
  const int mb = m0 + wv * 16 + (lane >> 4) * 8;
#pragma unroll
  for (int nt = 0; nt < 8; ++nt) {
    const int n = n0 + nt * 16 + (lane & 15);
    const float bb = bias[n];
#pragma unroll
    for (int v = 0; v < 8; ++v) {
      const int m = mb + v;
      const float val = acc[nt][v] + bb;
      if constexpr (MODE == 2) {
        out2[(size_t)m * NF_ + n] = val;
      } else {
        const int bI = m >> 12;          // m / T_
        const int t  = m & (T_ - 1);
        const int h  = n >> 6, d = n & 63;
        const size_t off = (((size_t)bI * H_ + h) * T_ + t) * DK_ + d;
        if constexpr (MODE == 0) {
          out0[off] = (__bf16)val;
        } else {
          out0[off] = (__bf16)(val + pbu[n]);
          out1[off] = (__bf16)(val + pbv[n]);
        }
      }
    }
  }
}

// ---------------------------------------------------------------------------
// p[h, j, d] = pos_emb[j] . Wp[h*64+d]  (j < 257, zero-padded to 272)
// ---------------------------------------------------------------------------
__global__ __launch_bounds__(64)
void pos_proj(const float* __restrict__ pe, const float* __restrict__ Wp,
              __bf16* __restrict__ pout) {
  const int j = blockIdx.x, d = threadIdx.x;
  if (j >= BAND_) {
#pragma unroll
    for (int h = 0; h < H_; ++h)
      pout[((size_t)h * BANDP_ + j) * DK_ + d] = (__bf16)0.0f;
    return;
  }
  for (int h = 0; h < H_; ++h) {
    const int n = h * DK_ + d;
    float s = 0.0f;
    for (int k = 0; k < NF_; ++k) s += pe[(size_t)j * NF_ + k] * Wp[(size_t)n * NF_ + k];
    pout[((size_t)h * BANDP_ + j) * DK_ + d] = (__bf16)s;
  }
}

// ---------------------------------------------------------------------------
// Banded attention; one workgroup (8 wave32) per (b, h, chunk of 128 rows).
// LDS (dynamic, 286720 B <= 320KB/WGP):
//   [0,139264)        : fp32 scores, 128 x 272
//   [139264, +98304)  : phase1 {q_v 16K | p 34K}  /  phase2 {q_u 16K | k_win 48K}
//                       phase4 : PA bf16 128x384 (96K)
//   [237568, +49152)  : v_win transposed (64 x 384) bf16
// Contiguous tiles (q_v, p, q_u) are streamed in with the Tensor Data Mover;
// windowed K/V need lower-OOB zero fill + transpose, so they stay on the
// vector path.
// ---------------------------------------------------------------------------
__global__ __launch_bounds__(256)
void attn_band_wmma(const __bf16* __restrict__ quG, const __bf16* __restrict__ qvG,
                    const __bf16* __restrict__ kG,  const __bf16* __restrict__ vG,
                    const __bf16* __restrict__ pG,
                    const unsigned char* __restrict__ padG,
                    float* __restrict__ ctx) {
  extern __shared__ char smem[];
  float*  sSc = (float*)smem;
  char*   r1  = smem + 139264;
  __bf16* sQV = (__bf16*)r1;
  __bf16* sP  = (__bf16*)(r1 + 16384);
  __bf16* sQU = (__bf16*)r1;
  __bf16* sKW = (__bf16*)(r1 + 16384);
  __bf16* sPA = (__bf16*)r1;
  __bf16* sVT = (__bf16*)(smem + 139264 + 98304);

  const int tid = threadIdx.x, wv = tid >> 5, lane = tid & 31;
  const int c = blockIdx.x, h = blockIdx.y, b = blockIdx.z;
  const int t0 = c * W_;
  const size_t headOff = ((size_t)b * H_ + h) * T_ * DK_;

  // ---- phase 0: stage q_v chunk + relative-pos table for this head --------
#if USE_TDM
  if (wv == 0) {
    tdm_load_1d((unsigned)(uintptr_t)(void*)sQV,
                qvG + headOff + (size_t)t0 * DK_, W_ * DK_ * 2);
    tdm_load_1d((unsigned)(uintptr_t)(void*)sP,
                pG + (size_t)h * BANDP_ * DK_, BANDP_ * DK_ * 2);
    __builtin_amdgcn_s_wait_tensorcnt(0);
  }
#else
  {
    const uint4* src = (const uint4*)(qvG + headOff + (size_t)t0 * DK_);
    uint4* dst = (uint4*)sQV;
    for (int i = tid; i < (W_ * DK_) / 8; i += 256) dst[i] = src[i];
    const uint4* ps = (const uint4*)(pG + (size_t)h * BANDP_ * DK_);
    uint4* pd = (uint4*)sP;
    for (int i = tid; i < (BANDP_ * DK_) / 8; i += 256) pd[i] = ps[i];
  }
#endif
  __syncthreads();

  // ---- phase 1: bd[x, j] = q_v @ p^T  (initializes the score band) --------
  {
    const __bf16* aB = sQV + wv * 16 * DK_;
    v16bf a0 = frag_a(aB, DK_);
    v16bf a1 = frag_a(aB + 32, DK_);
#pragma unroll 1
    for (int jt = 0; jt < 17; ++jt) {
      v8f acc = v8f_zero();
      acc = wmma32(a0, frag_b(sP + jt * 16 * DK_, DK_), acc);
      acc = wmma32(a1, frag_b(sP + jt * 16 * DK_ + 32, DK_), acc);
      const int mb = wv * 16 + (lane >> 4) * 8;
      const int n  = jt * 16 + (lane & 15);
#pragma unroll
      for (int v = 0; v < 8; ++v) sSc[(mb + v) * BANDP_ + n] = acc[v];
    }
  }
  __syncthreads();

  // ---- phase 2: stage q_u (TDM), windowed K (row-major), windowed V^T -----
#if USE_TDM
  if (wv == 0) {
    tdm_load_1d((unsigned)(uintptr_t)(void*)sQU,
                quG + headOff + (size_t)t0 * DK_, W_ * DK_ * 2);
  }
#else
  {
    const uint4* src = (const uint4*)(quG + headOff + (size_t)t0 * DK_);
    uint4* dst = (uint4*)sQU;
    for (int i = tid; i < (W_ * DK_) / 8; i += 256) dst[i] = src[i];
  }
#endif
  {
    const __bf16* kh = kG + headOff;
    const __bf16* vh = vG + headOff;
    for (int i = tid; i < WIN3_ * 8; i += 256) {
      const int y = i >> 3, seg = i & 7;
      const int ta = t0 - W_ + y;
      const bool ok = (ta >= 0) && (ta < T_);
      uint4 uk, uv;
      if (ok) {
        uk = *(const uint4*)(kh + (size_t)ta * DK_ + seg * 8);
        uv = *(const uint4*)(vh + (size_t)ta * DK_ + seg * 8);
      } else {
        uk.x = uk.y = uk.z = uk.w = 0u;
        uv.x = uv.y = uv.z = uv.w = 0u;
      }
      *(uint4*)(sKW + y * DK_ + seg * 8) = uk;
      union { uint4 q; __bf16 e[8]; } cv; cv.q = uv;
#pragma unroll
      for (int e = 0; e < 8; ++e) sVT[(seg * 8 + e) * WIN3_ + y] = cv.e[e];
    }
  }
#if USE_TDM
  if (wv == 0) __builtin_amdgcn_s_wait_tensorcnt(0);
#endif
  __syncthreads();

  // ---- phase 3: ac = q_u @ k_win^T, banded tiles, added into score band ---
  {
    const __bf16* aB = sQU + wv * 16 * DK_;
    v16bf a0 = frag_a(aB, DK_);
    v16bf a1 = frag_a(aB + 32, DK_);
#pragma unroll 1
    for (int u = 0; u < 17; ++u) {          // only diagonal-band N-tiles
      const int yt = wv + u;
      v8f acc = v8f_zero();
      acc = wmma32(a0, frag_b(sKW + yt * 16 * DK_, DK_), acc);
      acc = wmma32(a1, frag_b(sKW + yt * 16 * DK_ + 32, DK_), acc);
      const int mb = wv * 16 + (lane >> 4) * 8;
      const int yy = yt * 16 + (lane & 15);
#pragma unroll
      for (int v = 0; v < 8; ++v) {
        const int x = mb + v;
        const int j = yy - x;
        if (j >= 0 && j < BANDP_) sSc[x * BANDP_ + j] += acc[v];
      }
    }
  }
  __syncthreads();

  // ---- phase 4: masked softmax over the band (fp32, one thread per row) ---
  if (tid < W_) {
    const int x = tid;
    const int t_abs = t0 + x;
    float* row = sSc + x * BANDP_;
    const bool qpad = padG[(size_t)b * T_ + t_abs] != 0;
    float mx = -3.4e38f;
    for (int j = 0; j < BANDP_; ++j) {
      const int ka = t_abs + j - W_;
      float vl;
      if (j <= 256 && ka >= 0 && ka < T_) {
        vl = row[j] * 0.125f;                       // / sqrt(DK)=8
        if (padG[(size_t)b * T_ + ka]) vl += NEG_;
      } else {
        vl = NEG_;
      }
      row[j] = vl;
      mx = fmaxf(mx, vl);
    }
    float sum = 0.0f;
    for (int j = 0; j < BANDP_; ++j) {
      const float e = __expf(row[j] - mx);
      row[j] = e; sum += e;
    }
    const float inv = qpad ? 0.0f : (1.0f / sum);
    for (int j = 0; j < BANDP_; ++j) row[j] *= inv;
  }
  __syncthreads();

  // ---- phase 5: expand band probs -> dense bf16 PA (128 x 384) -----------
  for (int i = tid; i < W_ * WIN3_; i += 256) {
    const int x = i / WIN3_, y = i - x * WIN3_;
    const int j = y - x;
    const float vl = (j >= 0 && j <= 256) ? sSc[x * BANDP_ + j] : 0.0f;
    sPA[i] = (__bf16)vl;
  }
  __syncthreads();

  // ---- phase 6: out = PA @ v_win  (K = 384) -------------------------------
  {
    v8f acc[4];
#pragma unroll
    for (int i = 0; i < 4; ++i) acc[i] = v8f_zero();
#pragma unroll 1
    for (int ks = 0; ks < 12; ++ks) {
      v16bf a = frag_a(sPA + wv * 16 * WIN3_ + ks * 32, WIN3_);
#pragma unroll
      for (int nt = 0; nt < 4; ++nt)
        acc[nt] = wmma32(a, frag_b(sVT + nt * 16 * WIN3_ + ks * 32, WIN3_), acc[nt]);
    }
    const int mb = wv * 16 + (lane >> 4) * 8;
#pragma unroll
    for (int nt = 0; nt < 4; ++nt) {
      const int d = nt * 16 + (lane & 15);
#pragma unroll
      for (int v = 0; v < 8; ++v) {
        const int x = mb + v;
        ctx[((size_t)b * T_ + t0 + x) * NF_ + h * DK_ + d] = acc[nt][v];
      }
    }
  }
}

// ---------------------------------------------------------------------------
extern "C" void kernel_launch(void* const* d_in, const int* in_sizes, int n_in,
                              void* d_out, int out_size, void* d_ws, size_t ws_size,
                              hipStream_t stream) {
  (void)in_sizes; (void)n_in; (void)out_size; (void)ws_size;
  const float* query = (const float*)d_in[0];
  const float* key   = (const float*)d_in[1];
  const float* value = (const float*)d_in[2];
  const unsigned char* pad = (const unsigned char*)d_in[3];
  const float* pos_emb = (const float*)d_in[4];
  const float* Wq = (const float*)d_in[5];
  const float* bq = (const float*)d_in[6];
  const float* Wk = (const float*)d_in[7];
  const float* bk = (const float*)d_in[8];
  const float* Wv = (const float*)d_in[9];
  const float* bv = (const float*)d_in[10];
  const float* Wp = (const float*)d_in[11];
  const float* Wo = (const float*)d_in[12];
  const float* bo = (const float*)d_in[13];
  const float* pbu = (const float*)d_in[14];
  const float* pbv = (const float*)d_in[15];

  char* ws = (char*)d_ws;
  const size_t HEAD_BYTES = (size_t)B_ * H_ * T_ * DK_ * 2;    // 16 MB bf16
  const size_t P_BYTES    = (size_t)H_ * BANDP_ * DK_ * 2;     // 272 KB
  __bf16* qu  = (__bf16*)(ws);
  __bf16* qv  = (__bf16*)(ws + HEAD_BYTES);
  __bf16* kk  = (__bf16*)(ws + 2 * HEAD_BYTES);
  __bf16* vv  = (__bf16*)(ws + 3 * HEAD_BYTES);
  __bf16* pp  = (__bf16*)(ws + 4 * HEAD_BYTES);
  float*  ctx = (float*) (ws + 4 * HEAD_BYTES + P_BYTES);      // 32 MB fp32

  const dim3 gemmGrid(NF_ / 128, (B_ * T_) / 128);             // (4, 128)

  gemm512_wmma<1><<<gemmGrid, 256, 0, stream>>>(query, Wq, bq, qu, qv, nullptr, pbu, pbv);
  gemm512_wmma<0><<<gemmGrid, 256, 0, stream>>>(key,   Wk, bk, kk, nullptr, nullptr, nullptr, nullptr);
  gemm512_wmma<0><<<gemmGrid, 256, 0, stream>>>(value, Wv, bv, vv, nullptr, nullptr, nullptr, nullptr);
  pos_proj<<<BANDP_, 64, 0, stream>>>(pos_emb, Wp, pp);

  attn_band_wmma<<<dim3(T_ / W_, H_, B_), 256, 286720, stream>>>(
      qu, qv, kk, vv, pp, pad, ctx);

  gemm512_wmma<2><<<gemmGrid, 256, 0, stream>>>(ctx, Wo, bo, nullptr, nullptr,
                                                (float*)d_out, nullptr, nullptr);
}
